// EdgeTemporalSSM_41807211659523
// MI455X (gfx1250) — matrix operand order repaired and checked by reading, not compiled, not run on hardware
//
#include <hip/hip_runtime.h>
#include <hip/hip_bf16.h>

// EdgeTemporalSSM fused kernel for MI455X (gfx1250, wave32, WMMA).
//
// One workgroup per (b,e): GEMM1 (x = x_e@Wp^T+bp) -> LDS, GEMM2 (u = x@Wb^T+bb)
// -> LDS, sequential scan s_t = a*s_{t-1}+u_t in LDS, GEMM3+4
// (y = S@Wg^T + x@Wd^T + bg + bd), BN(eval) + exact GELU + residual x -> out.
// All GEMMs use v_wmma_f32_16x16x32_bf16 (bf16 inputs, f32 accumulate).

#define CIN   128
#define COUT  256
#define T_DIM 128
#define NBE   1024   // B*E = 4*256

typedef __attribute__((ext_vector_type(16))) __bf16        v16bf;
typedef __attribute__((ext_vector_type(8)))  float         v8f;
typedef __attribute__((ext_vector_type(8)))  unsigned int  v8u;

union FragU {
  v8u   u;
  v16bf b;
};

__device__ __forceinline__ unsigned short f2bf(float f) {
  unsigned int u = __float_as_uint(f);
  u += 0x7FFFu + ((u >> 16) & 1u);   // round-to-nearest-even
  return (unsigned short)(u >> 16);
}

__device__ __forceinline__ unsigned int cvt2_bf16(float lo, float hi) {
  unsigned int ul = __float_as_uint(lo);
  unsigned int uh = __float_as_uint(hi);
  ul += 0x7FFFu + ((ul >> 16) & 1u);
  uh += 0x7FFFu + ((uh >> 16) & 1u);
  return (ul >> 16) | (uh & 0xFFFF0000u);
}

// Build a 16x32 bf16 A-fragment for this lane from an f32 row pointer.
// A layout (16-bit, wave32), contiguous K to match B: lanes 0-15 (row M=lane):
// VGPR0..7 hold K=kstart+[0..16); lanes 16-31 (row M=lane-16):
// VGPR0..7 K=kstart+[16..32). Low 16 bits = lower K.
__device__ __forceinline__ FragU build_a_frag(const float* rowptr, int kstart, int hi) {
  FragU f;
  const float* p0 = rowptr + kstart + 16 * hi;
  const float* p1 = rowptr + kstart + 16 * hi + 8;
  float4 q0 = *(const float4*)(p0);
  float4 q1 = *(const float4*)(p0 + 4);
  float4 q2 = *(const float4*)(p1);
  float4 q3 = *(const float4*)(p1 + 4);
  f.u[0] = cvt2_bf16(q0.x, q0.y);
  f.u[1] = cvt2_bf16(q0.z, q0.w);
  f.u[2] = cvt2_bf16(q1.x, q1.y);
  f.u[3] = cvt2_bf16(q1.z, q1.w);
  f.u[4] = cvt2_bf16(q2.x, q2.y);
  f.u[5] = cvt2_bf16(q2.z, q2.w);
  f.u[6] = cvt2_bf16(q3.x, q3.y);
  f.u[7] = cvt2_bf16(q3.z, q3.w);
  return f;
}

// Load a 32x16 bf16 B-fragment: lane holds column n; lanes 0-15 cover
// K=kstart+[0..16), lanes 16-31 K=kstart+[16..32), 2 K-values per VGPR.
// wrow_u points at row n of the bf16 weight matrix viewed as uint (2 bf16/uint).
__device__ __forceinline__ FragU load_b_frag(const unsigned int* wrow_u, int khalf) {
  FragU f;
  uint4 a = *(const uint4*)(wrow_u + khalf);
  uint4 b = *(const uint4*)(wrow_u + khalf + 4);
  f.u[0] = a.x; f.u[1] = a.y; f.u[2] = a.z; f.u[3] = a.w;
  f.u[4] = b.x; f.u[5] = b.y; f.u[6] = b.z; f.u[7] = b.w;
  return f;
}

// ---------- precompute: weights f32 -> bf16, fold tanh(a) and BN params ----------
__global__ __launch_bounds__(256) void edgessm_precompute(
    const float* __restrict__ Wp, const float* __restrict__ Wb,
    const float* __restrict__ Wg, const float* __restrict__ Wd,
    const float* __restrict__ a_param,
    const float* __restrict__ gamma, const float* __restrict__ beta,
    const float* __restrict__ run_mean, const float* __restrict__ run_var,
    unsigned short* __restrict__ wp_bf, unsigned short* __restrict__ wb_bf,
    unsigned short* __restrict__ wg_bf, unsigned short* __restrict__ wd_bf,
    float* __restrict__ a_tanh, float* __restrict__ bn_scale, float* __restrict__ bn_shift) {
  int idx = blockIdx.x * blockDim.x + threadIdx.x;
  const int NWP = COUT * CIN;    // 32768
  const int NW  = COUT * COUT;   // 65536
  if (idx < NWP) wp_bf[idx] = f2bf(Wp[idx]);
  if (idx < NW) {
    wb_bf[idx] = f2bf(Wb[idx]);
    wg_bf[idx] = f2bf(Wg[idx]);
    wd_bf[idx] = f2bf(Wd[idx]);
  }
  if (idx < COUT) {
    a_tanh[idx] = tanhf(a_param[idx]);
    float sc = gamma[idx] * rsqrtf(run_var[idx] + 1e-5f);
    bn_scale[idx] = sc;
    bn_shift[idx] = beta[idx] - run_mean[idx] * sc;
  }
}

// ---------- fused main kernel: one workgroup per (b,e) ----------
__global__ __launch_bounds__(256) void edgessm_main(
    const float* __restrict__ x_e,
    const float* __restrict__ bp, const float* __restrict__ bb,
    const float* __restrict__ bg, const float* __restrict__ bd,
    const unsigned short* __restrict__ wp_bf, const unsigned short* __restrict__ wb_bf,
    const unsigned short* __restrict__ wg_bf, const unsigned short* __restrict__ wd_bf,
    const float* __restrict__ a_tanh, const float* __restrict__ bn_scale,
    const float* __restrict__ bn_shift,
    float* __restrict__ out) {
  extern __shared__ float smem[];
  float* ldsX = smem;                   // [T_DIM][COUT] f32: x (GEMM-A for Wd + residual)
  float* ldsU = smem + T_DIM * COUT;    // [T_DIM][COUT] f32: u, then S in place

  const int be   = blockIdx.x;
  const int tid  = threadIdx.x;
  const int wave = tid >> 5;
  const int lane = tid & 31;
  const int lm   = lane & 15;   // col within tile (C/D, B layouts) / row (A layout)
  const int hi   = lane >> 4;

  const int mbase = wave * 16;  // wave owns M-tile rows [mbase, mbase+16)

  // ===== Stage 1: x = x_e @ Wp^T + bp  -> ldsX =====
  const float* arow_g = x_e + ((size_t)be * T_DIM + (mbase + lm)) * CIN;
  FragU a1[4];
#pragma unroll
  for (int kf = 0; kf < 4; ++kf) a1[kf] = build_a_frag(arow_g, kf * 32, hi);

#pragma unroll 1
  for (int nt = 0; nt < 16; ++nt) {
    const int n = nt * 16 + lm;
    float bias = bp[n];
    v8f acc;
#pragma unroll
    for (int i = 0; i < 8; ++i) acc[i] = bias;
    const unsigned int* wrow = (const unsigned int*)(wp_bf + (size_t)n * CIN);
#pragma unroll
    for (int kf = 0; kf < 4; ++kf) {
      FragU bf = load_b_frag(wrow, (kf * 32 + 16 * hi) >> 1);
      acc = __builtin_amdgcn_wmma_f32_16x16x32_bf16(false, a1[kf].b, false, bf.b,
                                                    (short)0, acc, false, false);
    }
#pragma unroll
    for (int i = 0; i < 8; ++i) {
      int row = mbase + i + 8 * hi;          // C layout: VGPR i -> M=i (+8 for hi lanes)
      ldsX[row * COUT + n] = acc[i];
    }
  }
  __syncthreads();

  // ===== Stage 2: u = x @ Wb^T + bb  -> ldsU =====
  const float* xrow = ldsX + (mbase + lm) * COUT;
  FragU ax[8];                                // kept live through stage 4 (Wd GEMM)
#pragma unroll
  for (int kf = 0; kf < 8; ++kf) ax[kf] = build_a_frag(xrow, kf * 32, hi);

#pragma unroll 1
  for (int nt = 0; nt < 16; ++nt) {
    const int n = nt * 16 + lm;
    float bias = bb[n];
    v8f acc;
#pragma unroll
    for (int i = 0; i < 8; ++i) acc[i] = bias;
    const unsigned int* wrow = (const unsigned int*)(wb_bf + (size_t)n * COUT);
#pragma unroll
    for (int kf = 0; kf < 8; ++kf) {
      FragU bf = load_b_frag(wrow, (kf * 32 + 16 * hi) >> 1);
      acc = __builtin_amdgcn_wmma_f32_16x16x32_bf16(false, ax[kf].b, false, bf.b,
                                                    (short)0, acc, false, false);
    }
#pragma unroll
    for (int i = 0; i < 8; ++i) {
      int row = mbase + i + 8 * hi;
      ldsU[row * COUT + n] = acc[i];
    }
  }
  __syncthreads();

  // ===== Stage 3: scan s_t = a*s_{t-1} + u_t  (thread tid owns channel tid) =====
  {
    float a = a_tanh[tid];
    float s = 0.0f;
#pragma unroll 4
    for (int t = 0; t < T_DIM; ++t) {
      int idx = t * COUT + tid;              // consecutive lanes -> conflict-free banks
      s = fmaf(a, s, ldsU[idx]);
      ldsU[idx] = s;
    }
  }
  __syncthreads();

  // ===== Stage 4: y = S@Wg^T + x@Wd^T + (bg+bd); BN + GELU + residual -> out =====
  const float* srow = ldsU + (mbase + lm) * COUT;
  FragU as[8];
#pragma unroll
  for (int kf = 0; kf < 8; ++kf) as[kf] = build_a_frag(srow, kf * 32, hi);

  const size_t outbase = (size_t)be * T_DIM * COUT;
#pragma unroll 1
  for (int nt = 0; nt < 16; ++nt) {
    const int n = nt * 16 + lm;
    float bias = bg[n] + bd[n];
    v8f acc;
#pragma unroll
    for (int i = 0; i < 8; ++i) acc[i] = bias;
    const unsigned int* wgrow = (const unsigned int*)(wg_bf + (size_t)n * COUT);
    const unsigned int* wdrow = (const unsigned int*)(wd_bf + (size_t)n * COUT);
#pragma unroll
    for (int kf = 0; kf < 8; ++kf) {
      int khalf = (kf * 32 + 16 * hi) >> 1;
      FragU bfg = load_b_frag(wgrow, khalf);
      acc = __builtin_amdgcn_wmma_f32_16x16x32_bf16(false, as[kf].b, false, bfg.b,
                                                    (short)0, acc, false, false);
      FragU bfd = load_b_frag(wdrow, khalf);
      acc = __builtin_amdgcn_wmma_f32_16x16x32_bf16(false, ax[kf].b, false, bfd.b,
                                                    (short)0, acc, false, false);
    }
    const float scale = bn_scale[n];
    const float shift = bn_shift[n];
#pragma unroll
    for (int i = 0; i < 8; ++i) {
      int row = mbase + i + 8 * hi;
      float y = acc[i] * scale + shift;
      float g = 0.5f * y * (1.0f + erff(y * 0.70710678118654752440f));  // exact GELU
      out[outbase + (size_t)row * COUT + n] = g + ldsX[row * COUT + n];
    }
  }
}

extern "C" void kernel_launch(void* const* d_in, const int* in_sizes, int n_in,
                              void* d_out, int out_size, void* d_ws, size_t ws_size,
                              hipStream_t stream) {
  const float* x_e   = (const float*)d_in[0];
  const float* Wp    = (const float*)d_in[1];
  const float* bp    = (const float*)d_in[2];
  const float* a_par = (const float*)d_in[3];
  const float* Wb    = (const float*)d_in[4];
  const float* bb    = (const float*)d_in[5];
  const float* Wg    = (const float*)d_in[6];
  const float* bg    = (const float*)d_in[7];
  const float* Wd    = (const float*)d_in[8];
  const float* bd    = (const float*)d_in[9];
  const float* gamma = (const float*)d_in[10];
  const float* beta  = (const float*)d_in[11];
  const float* rmean = (const float*)d_in[12];
  const float* rvar  = (const float*)d_in[13];

  char* ws = (char*)d_ws;
  unsigned short* wp_bf = (unsigned short*)(ws);                        // 256*128*2
  unsigned short* wb_bf = (unsigned short*)(ws + 65536);                // 256*256*2
  unsigned short* wg_bf = (unsigned short*)(ws + 65536 + 131072);
  unsigned short* wd_bf = (unsigned short*)(ws + 65536 + 2 * 131072);
  float* a_tanh   = (float*)(ws + 65536 + 3 * 131072);
  float* bn_scale = a_tanh + COUT;
  float* bn_shift = bn_scale + COUT;

  edgessm_precompute<<<256, 256, 0, stream>>>(
      Wp, Wb, Wg, Wd, a_par, gamma, beta, rmean, rvar,
      wp_bf, wb_bf, wg_bf, wd_bf, a_tanh, bn_scale, bn_shift);

  const size_t lds_bytes = 2u * T_DIM * COUT * sizeof(float);  // 256 KB (<=320 KB WGP LDS)
  hipFuncSetAttribute((const void*)edgessm_main,
                      hipFuncAttributeMaxDynamicSharedMemorySize, (int)lds_bytes);
  edgessm_main<<<NBE, 256, lds_bytes, stream>>>(
      x_e, bp, bb, bg, bd, wp_bf, wb_bf, wg_bf, wd_bf,
      a_tanh, bn_scale, bn_shift, (float*)d_out);
}
